// ValueRNN_10728828305615
// MI455X (gfx1250) — matrix operand order repaired and checked by reading, not compiled, not run on hardware
//
#include <hip/hip_runtime.h>
#include <hip/hip_bf16.h>

// ValueRNN (GRU, T=512 B=256 I=64 H=512 O=1) for MI455X / gfx1250.
// Strategy:
//  - prep kernel: f32->f16 convert X, W_ih, W_hh; zero h ping-pong; out = bias.
//  - per-timestep fused kernel: WMMA f32_16x16x32_f16 for X-proj (K=64) and
//    h-proj (K=512), gate nonlinearities, writes h_{t+1} (f16, next step's A),
//    and fuses the O=1 value projection via shfl-reduce + atomicAdd.
// Workspace (f16): X 8.4M + Wih 98K + Whh 786K + 2*h 262K  ~= 19.1 MB.

typedef _Float16 v8h  __attribute__((ext_vector_type(8)));
typedef _Float16 v16h __attribute__((ext_vector_type(16)));
typedef float    v8f  __attribute__((ext_vector_type(8)));

constexpr int T = 512, B = 256, I = 64, H = 512;

// ---- WMMA helpers -----------------------------------------------------------

__device__ __forceinline__ v8f wmma_f16(v16h a, v16h b, v8f c) {
  // (neg_a, A, neg_b, B, c_mod, C, reuse_a, reuse_b)
  return __builtin_amdgcn_wmma_f32_16x16x32_f16(false, a, false, b,
                                                (short)0, c, false, false);
}

// A-matrix 16x32 f16 layout (ISA 7.12.2): lane grp=lane>>4 holds
// K = kk + grp*8 + {0..7}  and  K = kk + 16 + grp*8 + {0..7}.
__device__ __forceinline__ v16h load_a(const _Float16* __restrict__ rowp,
                                       int kk, int grp) {
  v8h lo = *(const v8h*)(rowp + kk + grp * 8);
  v8h hi = *(const v8h*)(rowp + kk + 16 + grp * 8);
  return __builtin_shufflevector(lo, hi, 0, 1, 2, 3, 4, 5, 6, 7,
                                 8, 9, 10, 11, 12, 13, 14, 15);
}

// B-matrix 32x16 f16: lane holds column N=lane&15, K = kk + grp*8 + {0..7}
// and K = kk + 16 + grp*8 + {0..7} (same K order as A; W stored (N,K) row-major).
__device__ __forceinline__ v16h load_b(const _Float16* __restrict__ rowp,
                                       int kk, int grp) {
  v8h lo = *(const v8h*)(rowp + kk + grp * 8);
  v8h hi = *(const v8h*)(rowp + kk + 16 + grp * 8);
  return __builtin_shufflevector(lo, hi, 0, 1, 2, 3, 4, 5, 6, 7,
                                 8, 9, 10, 11, 12, 13, 14, 15);
}

__device__ __forceinline__ float fsigmoid(float x) {
  return 1.0f / (1.0f + __expf(-x));
}
__device__ __forceinline__ float ftanh(float x) {
  x = fminf(15.0f, fmaxf(-15.0f, x));
  float e2 = __expf(2.0f * x);
  return (e2 - 1.0f) / (e2 + 1.0f);
}

// ---- prep: convert to f16, zero h, init output with bias --------------------

__global__ void valuernn_prep_kernel(const float* __restrict__ X,
                                     const float* __restrict__ wih,
                                     const float* __restrict__ whh,
                                     const float* __restrict__ bias0,
                                     _Float16* __restrict__ Xh,
                                     _Float16* __restrict__ Wih,
                                     _Float16* __restrict__ Whh,
                                     _Float16* __restrict__ hbuf,
                                     float* __restrict__ out) {
  const int i = blockIdx.x * blockDim.x + threadIdx.x;
  if (i < T * B * I)   Xh[i]   = (_Float16)X[i];
  if (i < 3 * H * I)   Wih[i]  = (_Float16)wih[i];
  if (i < 3 * H * H)   Whh[i]  = (_Float16)whh[i];
  if (i < 2 * B * H)   hbuf[i] = (_Float16)0.0f;
  if (i < T * B)       out[i]  = bias0[0];
}

// ---- one GRU timestep, fully fused ------------------------------------------
// Grid: 256 blocks x 64 threads (2 waves/block) = 512 waves = 16x32 tiles
// covering the (B=256) x (H=512) hidden-state update.

__global__ __launch_bounds__(64)
void valuernn_gru_step_kernel(const _Float16* __restrict__ Xh,   // (T,B,I)
                              const _Float16* __restrict__ Wih,  // (3H,I)
                              const _Float16* __restrict__ Whh,  // (3H,H)
                              const float* __restrict__ bih,     // (3H)
                              const float* __restrict__ bhh,     // (3H)
                              const float* __restrict__ vw,      // (H)
                              const _Float16* __restrict__ hprev,// (B,H)
                              _Float16* __restrict__ hnext,      // (B,H)
                              float* __restrict__ out,           // (T,B)
                              int t) {
  const int lane = threadIdx.x & 31;
  const int wave = threadIdx.x >> 5;
  const int tile = blockIdx.x * 2 + wave;     // [0, 512)
  const int m0 = (tile & 15) * 16;            // batch-tile origin
  const int n0 = (tile >> 4) * 16;            // hidden-tile origin
  const int row = lane & 15;
  const int grp = lane >> 4;

  v8f c_r  = {}; // reset gate:  x-proj + h-proj
  v8f c_z  = {}; // update gate: x-proj + h-proj
  v8f c_nx = {}; // new gate, x part (r is NOT applied to this)
  v8f c_nh = {}; // new gate, h part (r IS applied to this)

  // Prefetch next step's X tile into cache (global_prefetch_b8 path).
  if (t + 1 < T)
    __builtin_prefetch(Xh + ((size_t)(t + 1) * B + m0 + row) * I, 0, 1);

  // --- X projection: K = I = 64 (2 WMMA k-chunks x 3 gates) ---
  {
    const _Float16* ax  = Xh + ((size_t)t * B + m0 + row) * I;
    const _Float16* bwr = Wih + (size_t)(0 * H + n0 + row) * I;
    const _Float16* bwz = Wih + (size_t)(1 * H + n0 + row) * I;
    const _Float16* bwn = Wih + (size_t)(2 * H + n0 + row) * I;
#pragma unroll
    for (int kk = 0; kk < I; kk += 32) {
      v16h a = load_a(ax, kk, grp);
      c_r  = wmma_f16(a, load_b(bwr, kk, grp), c_r);
      c_z  = wmma_f16(a, load_b(bwz, kk, grp), c_z);
      c_nx = wmma_f16(a, load_b(bwn, kk, grp), c_nx);
    }
  }

  // --- h projection: K = H = 512 (16 WMMA k-chunks x 3 gates) ---
  {
    const _Float16* ah  = hprev + (size_t)(m0 + row) * H;
    const _Float16* bwr = Whh + (size_t)(0 * H + n0 + row) * H;
    const _Float16* bwz = Whh + (size_t)(1 * H + n0 + row) * H;
    const _Float16* bwn = Whh + (size_t)(2 * H + n0 + row) * H;
#pragma unroll 4
    for (int kk = 0; kk < H; kk += 32) {
      v16h a = load_a(ah, kk, grp);
      c_r  = wmma_f16(a, load_b(bwr, kk, grp), c_r);
      c_z  = wmma_f16(a, load_b(bwz, kk, grp), c_z);
      c_nh = wmma_f16(a, load_b(bwn, kk, grp), c_nh);
    }
  }

  // --- gates + state update + fused O=1 value projection ---
  const int col = n0 + row;                   // hidden index (C/D: N = lane&15)
  const float bir = bih[col],         bhr = bhh[col];
  const float biz = bih[H + col],     bhz = bhh[H + col];
  const float bin = bih[2 * H + col], bhn = bhh[2 * H + col];
  const float w   = vw[col];

#pragma unroll
  for (int r2 = 0; r2 < 8; ++r2) {
    const int brow = m0 + r2 + grp * 8;       // C/D row: M = vgpr + 8*(lane>>4)
    const float hp = (float)hprev[(size_t)brow * H + col];
    const float rg = fsigmoid(c_r[r2] + bir + bhr);
    const float zg = fsigmoid(c_z[r2] + biz + bhz);
    const float ng = ftanh(c_nx[r2] + bin + rg * (c_nh[r2] + bhn));
    const float hn = (1.0f - zg) * ng + zg * hp;
    hnext[(size_t)brow * H + col] = (_Float16)hn;

    // out[t][brow] += sum_col hn * vw[col]; reduce over the 16-lane N group.
    float p = hn * w;
    p += __shfl_xor(p, 1);
    p += __shfl_xor(p, 2);
    p += __shfl_xor(p, 4);
    p += __shfl_xor(p, 8);
    if (row == 0) atomicAdd(&out[(size_t)t * B + brow], p);
  }
}

// ---- host-side launcher -----------------------------------------------------

extern "C" void kernel_launch(void* const* d_in, const int* in_sizes, int n_in,
                              void* d_out, int out_size, void* d_ws, size_t ws_size,
                              hipStream_t stream) {
  const float* X    = (const float*)d_in[0];  // (T,B,I)
  const float* wih  = (const float*)d_in[1];  // (3H,I)
  const float* whh  = (const float*)d_in[2];  // (3H,H)
  const float* bih  = (const float*)d_in[3];  // (3H)
  const float* bhh  = (const float*)d_in[4];  // (3H)
  const float* vw   = (const float*)d_in[5];  // (O=1,H)
  const float* bias = (const float*)d_in[6];  // (1)
  float* out = (float*)d_out;                 // (T,B,1)

  // Workspace carve-up (all f16, every segment 16B-aligned).
  _Float16* ws  = (_Float16*)d_ws;
  _Float16* Xh  = ws;                         // T*B*I   = 8,388,608
  _Float16* Wih = Xh  + (size_t)T * B * I;    // 3H*I    = 98,304
  _Float16* Whh = Wih + (size_t)3 * H * I;    // 3H*H    = 786,432
  _Float16* hbf = Whh + (size_t)3 * H * H;    // 2*B*H   = 262,144
  // total: ~19.1 MB of ws_size

  {
    const int n = T * B * I;                  // largest conversion range
    valuernn_prep_kernel<<<(n + 255) / 256, 256, 0, stream>>>(
        X, wih, whh, bias, Xh, Wih, Whh, hbf, out);
  }

  for (int t = 0; t < T; ++t) {
    const _Float16* hp = hbf + (size_t)(t & 1) * B * H;
    _Float16*       hn = hbf + (size_t)((t + 1) & 1) * B * H;
    valuernn_gru_step_kernel<<<256, 64, 0, stream>>>(
        Xh, Wih, Whh, bih, bhh, vw, hp, hn, out, t);
  }
}
